// MHSA_84945863180702
// MI455X (gfx1250) — compile-verified, matrix operand order checked by reading, not run on hardware
//
#include <hip/hip_runtime.h>

// ---------------------------------------------------------------------------
// MHSA (B=1, N=4096, D=768, H=12, hd=64) for gfx1250 (MI455X).
// ~71 GFLOP vs ~25 MB of unique traffic (everything L2-resident) -> hard
// compute-bound -> all matmuls on v_wmma_f32_16x16x32_bf16 (fp32 accum),
// fp32 softmax. Fragment loads double-buffered so VMEM overlaps the WMMA
// pipe instead of s_wait_loadcnt 0 before every matrix op.
// ---------------------------------------------------------------------------

typedef __attribute__((ext_vector_type(16))) __bf16        v16bf;
typedef __attribute__((ext_vector_type(8)))  float         v8f;
typedef __attribute__((ext_vector_type(4)))  unsigned int  v4u;

union FragBF {
    v4u   q[2];   // two 16-byte halves (A/B fragment = 32 bytes / lane)
    v16bf v;
};

static __device__ __forceinline__ void loadFrag(FragBF& f, const unsigned short* p) {
    const v4u* q = (const v4u*)p;
    f.q[0] = q[0];
    f.q[1] = q[2];   // +16 bf16 elements (second 16-byte half)
}

static __device__ __forceinline__ unsigned short f2bf(float f) {
    unsigned int u = __float_as_uint(f);
    u += 0x7FFFu + ((u >> 16) & 1u);        // round-to-nearest-even
    return (unsigned short)(u >> 16);
}

constexpr int EMBED = 768;
constexpr int SEQ   = 4096;
constexpr int HEADS = 12;
constexpr int HD    = 64;

// ---------------------------------------------------------------------------
// fp32 -> bf16 downconvert (one-time, bandwidth-trivial)
// ---------------------------------------------------------------------------
__global__ void cvt_f32_bf16(const float* __restrict__ in,
                             unsigned short* __restrict__ out, int n) {
    int i = blockIdx.x * blockDim.x + threadIdx.x;
    if (i < n) out[i] = f2bf(in[i]);
}

// ---------------------------------------------------------------------------
// QKV projection: qkv[m][n] = sum_k x[m][k] * w_qkv[n][k] + b_qkv[n]
// 8 waves/block, each wave owns a 32(M) x 64(N) tile: 8 wmma per 32-wide
// k-chunk against 6 b128 fragment loads, double-buffered across chunks.
// Stores attention-friendly layouts: Q [h][n][64] (pre-scaled 1/8),
// K [h][n][64], V transposed [h][64][n].
// ---------------------------------------------------------------------------
__global__ __launch_bounds__(256) void qkv_gemm_wmma(
    const unsigned short* __restrict__ xbf,   // [SEQ][EMBED] bf16
    const unsigned short* __restrict__ wbf,   // [3*EMBED][EMBED] bf16
    const float*          __restrict__ bias,  // [3*EMBED]
    unsigned short* __restrict__ qout,        // [HEADS][SEQ][HD]
    unsigned short* __restrict__ kout,        // [HEADS][SEQ][HD]
    unsigned short* __restrict__ vtout)       // [HEADS][HD][SEQ]
{
    const int K     = EMBED;
    const int lane  = threadIdx.x & 31;
    const int wv    = threadIdx.x >> 5;
    const int l15   = lane & 15;
    const int hhalf = lane >> 4;
    const int mBase = blockIdx.y * 256 + wv * 32;
    const int nBase = blockIdx.x * 64;

    const unsigned short* arow = xbf + (size_t)(mBase + l15) * K + hhalf * 8;
    const unsigned short* brow = wbf + (size_t)(nBase + l15) * K + hhalf * 8;

    v8f acc[2][4] = {};
    FragBF a0[2], b0[4], a1[2], b1[4];

    auto loadAB = [&](FragBF* a, FragBF* b, int kc) {
#pragma unroll
        for (int mf = 0; mf < 2; ++mf)
            loadFrag(a[mf], arow + (size_t)mf * 16 * K + kc);
#pragma unroll
        for (int f = 0; f < 4; ++f)
            loadFrag(b[f], brow + (size_t)f * 16 * K + kc);
    };
    auto compute = [&](FragBF* a, FragBF* b) {
#pragma unroll
        for (int mf = 0; mf < 2; ++mf)
#pragma unroll
            for (int f = 0; f < 4; ++f)
                acc[mf][f] = __builtin_amdgcn_wmma_f32_16x16x32_bf16(
                    false, a[mf].v, false, b[f].v, (short)0, acc[mf][f],
                    false, false);
    };

    loadAB(a0, b0, 0);
    int kc = 0;
    for (; kc + 128 <= K; kc += 64) {           // software-pipelined mainloop
        __builtin_prefetch(arow + kc + 128, 0, 1);   // -> global_prefetch_b8
        __builtin_prefetch(brow + kc + 128, 0, 1);
        loadAB(a1, b1, kc + 32);
        compute(a0, b0);
        loadAB(a0, b0, kc + 64);
        compute(a1, b1);
    }
    loadAB(a1, b1, kc + 32);                    // kc == K-64 here
    compute(a0, b0);
    compute(a1, b1);

#pragma unroll
    for (int f = 0; f < 4; ++f) {
        const int ncol  = nBase + f * 16 + l15;
        const int which = ncol / EMBED;            // 0=q 1=k 2=v
        const int rem   = ncol - which * EMBED;
        const int head  = rem >> 6;
        const int hd    = rem & 63;
        const float bv  = bias[ncol];
#pragma unroll
        for (int mf = 0; mf < 2; ++mf)
#pragma unroll
            for (int r = 0; r < 8; ++r) {
                const int   m   = mBase + mf * 16 + r + hhalf * 8;
                const float val = acc[mf][f][r] + bv;
                if (which == 0) {
                    qout[((size_t)head * SEQ + m) * HD + hd] = f2bf(val * 0.125f);
                } else if (which == 1) {
                    kout[((size_t)head * SEQ + m) * HD + hd] = f2bf(val);
                } else {
                    vtout[((size_t)head * HD + hd) * SEQ + m] = f2bf(val);
                }
            }
    }
}

// ---------------------------------------------------------------------------
// Flash-attention: 4 waves/block, wave owns 16 query rows of one head.
// Per 64-key block: batch-load all K and V fragments (V latency hides under
// the fp32 softmax VALU section), 8 wmma QK^T + online softmax + 8 wmma P.V.
// P re-layout C-frag -> A-frag goes through a per-wave 2KB LDS tile.
// ---------------------------------------------------------------------------
__global__ __launch_bounds__(128) void attn_wmma(
    const unsigned short* __restrict__ qbf,   // [HEADS][SEQ][HD] (pre-scaled)
    const unsigned short* __restrict__ kbf,   // [HEADS][SEQ][HD]
    const unsigned short* __restrict__ vtbf,  // [HEADS][HD][SEQ]
    unsigned short* __restrict__ attnbf)      // [SEQ][EMBED]
{
    const int lane  = threadIdx.x & 31;
    const int wv    = threadIdx.x >> 5;
    const int l15   = lane & 15;
    const int hhalf = lane >> 4;
    const int head  = blockIdx.y;
    const int q0    = blockIdx.x * 64 + wv * 16;

    __shared__ __align__(16) unsigned short pbuf[4][16 * 64];  // 8 KB
    unsigned short* myp = pbuf[wv];

    FragBF aq[2];
    {
        const unsigned short* qrow =
            qbf + ((size_t)head * SEQ + q0 + l15) * HD + hhalf * 8;
#pragma unroll
        for (int kc = 0; kc < 2; ++kc) loadFrag(aq[kc], qrow + kc * 32);
    }

    float mrow[8], lrow[8];
#pragma unroll
    for (int r = 0; r < 8; ++r) { mrow[r] = -3.0e38f; lrow[r] = 0.0f; }
    v8f o[4] = {};

    for (int jb = 0; jb < SEQ; jb += 64) {
        // ---- batch-load K and V fragments for this key block ----
        FragBF kF[4][2], vF[4][2];
#pragma unroll
        for (int nf = 0; nf < 4; ++nf) {
            const unsigned short* krow =
                kbf + ((size_t)head * SEQ + jb + nf * 16 + l15) * HD + hhalf * 8;
#pragma unroll
            for (int kc = 0; kc < 2; ++kc) loadFrag(kF[nf][kc], krow + kc * 32);
        }
#pragma unroll
        for (int nf = 0; nf < 4; ++nf) {
            const unsigned short* vrow =
                vtbf + ((size_t)head * HD + nf * 16 + l15) * SEQ + jb + hhalf * 8;
#pragma unroll
            for (int kc = 0; kc < 2; ++kc) loadFrag(vF[nf][kc], vrow + kc * 32);
        }
        if (jb + 64 < SEQ) {   // pull next key block toward the caches
            __builtin_prefetch(
                kbf + ((size_t)head * SEQ + jb + 64 + l15) * HD, 0, 1);
            __builtin_prefetch(
                vtbf + ((size_t)head * HD + l15) * SEQ + jb + 64, 0, 1);
        }

        // ---- S = Q . K^T for a 16x64 tile ----
        v8f s[4] = {};
#pragma unroll
        for (int nf = 0; nf < 4; ++nf)
#pragma unroll
            for (int kc = 0; kc < 2; ++kc)
                s[nf] = __builtin_amdgcn_wmma_f32_16x16x32_bf16(
                    false, aq[kc].v, false, kF[nf][kc].v, (short)0, s[nf],
                    false, false);

        // ---- online softmax (rows live across 16-lane halves) ----
        float nm[8];
#pragma unroll
        for (int r = 0; r < 8; ++r)
            nm[r] = fmaxf(fmaxf(s[0][r], s[1][r]), fmaxf(s[2][r], s[3][r]));
#pragma unroll
        for (int msk = 1; msk <= 8; msk <<= 1)
#pragma unroll
            for (int r = 0; r < 8; ++r)
                nm[r] = fmaxf(nm[r], __shfl_xor(nm[r], msk, 32));

        float sc[8], rs[8];
#pragma unroll
        for (int r = 0; r < 8; ++r) {
            const float mn = fmaxf(mrow[r], nm[r]);
            sc[r]   = __expf(mrow[r] - mn);
            mrow[r] = mn;
            rs[r]   = 0.0f;
        }
#pragma unroll
        for (int nf = 0; nf < 4; ++nf)
#pragma unroll
            for (int r = 0; r < 8; ++r) {
                const float pv = __expf(s[nf][r] - mrow[r]);
                s[nf][r] = pv;
                rs[r] += pv;
            }
#pragma unroll
        for (int msk = 1; msk <= 8; msk <<= 1)
#pragma unroll
            for (int r = 0; r < 8; ++r)
                rs[r] += __shfl_xor(rs[r], msk, 32);
#pragma unroll
        for (int r = 0; r < 8; ++r)
            lrow[r] = lrow[r] * sc[r] + rs[r];
#pragma unroll
        for (int f = 0; f < 4; ++f)
#pragma unroll
            for (int r = 0; r < 8; ++r)
                o[f][r] *= sc[r];

        // ---- spill P (bf16) to per-wave LDS tile: C-frag -> row-major ----
#pragma unroll
        for (int nf = 0; nf < 4; ++nf)
#pragma unroll
            for (int r = 0; r < 8; ++r) {
                const int qq  = r + hhalf * 8;
                const int key = nf * 16 + l15;
                myp[qq * 64 + key] = f2bf(s[nf][r]);
            }
        asm volatile("s_wait_dscnt 0x0" ::: "memory");  // DS RAW, same wave

        // ---- reload P as A-fragments (16 x 64 over two k-chunks) ----
        FragBF pa[2];
        {
            const unsigned short* prow = myp + l15 * 64 + hhalf * 8;
#pragma unroll
            for (int kc = 0; kc < 2; ++kc) loadFrag(pa[kc], prow + kc * 32);
        }

        // ---- O += P . V (V fragments already resident in registers) ----
#pragma unroll
        for (int nf = 0; nf < 4; ++nf)
#pragma unroll
            for (int kc = 0; kc < 2; ++kc)
                o[nf] = __builtin_amdgcn_wmma_f32_16x16x32_bf16(
                    false, pa[kc].v, false, vF[nf][kc].v, (short)0, o[nf],
                    false, false);
    }

    // ---- normalize and emit bf16 [SEQ][EMBED] for the out-projection ----
#pragma unroll
    for (int f = 0; f < 4; ++f)
#pragma unroll
        for (int r = 0; r < 8; ++r) {
            const int m   = q0 + r + hhalf * 8;
            const int col = head * HD + f * 16 + l15;
            attnbf[(size_t)m * EMBED + col] = f2bf(o[f][r] / lrow[r]);
        }
}

// ---------------------------------------------------------------------------
// Output projection: out[m][n] = sum_k attn[m][k] * w_out[n][k] + b_out[n]
// Same double-buffered 32x64-per-wave skeleton, fp32 output + bias.
// ---------------------------------------------------------------------------
__global__ __launch_bounds__(256) void out_gemm_wmma(
    const unsigned short* __restrict__ abf,   // [SEQ][EMBED] bf16
    const unsigned short* __restrict__ wbf,   // [EMBED][EMBED] bf16
    const float*          __restrict__ bias,  // [EMBED]
    float* __restrict__ out)                  // [SEQ][EMBED] fp32
{
    const int K     = EMBED;
    const int lane  = threadIdx.x & 31;
    const int wv    = threadIdx.x >> 5;
    const int l15   = lane & 15;
    const int hhalf = lane >> 4;
    const int mBase = blockIdx.y * 256 + wv * 32;
    const int nBase = blockIdx.x * 64;

    const unsigned short* arow = abf + (size_t)(mBase + l15) * K + hhalf * 8;
    const unsigned short* brow = wbf + (size_t)(nBase + l15) * K + hhalf * 8;

    v8f acc[2][4] = {};
    FragBF a0[2], b0[4], a1[2], b1[4];

    auto loadAB = [&](FragBF* a, FragBF* b, int kc) {
#pragma unroll
        for (int mf = 0; mf < 2; ++mf)
            loadFrag(a[mf], arow + (size_t)mf * 16 * K + kc);
#pragma unroll
        for (int f = 0; f < 4; ++f)
            loadFrag(b[f], brow + (size_t)f * 16 * K + kc);
    };
    auto compute = [&](FragBF* a, FragBF* b) {
#pragma unroll
        for (int mf = 0; mf < 2; ++mf)
#pragma unroll
            for (int f = 0; f < 4; ++f)
                acc[mf][f] = __builtin_amdgcn_wmma_f32_16x16x32_bf16(
                    false, a[mf].v, false, b[f].v, (short)0, acc[mf][f],
                    false, false);
    };

    loadAB(a0, b0, 0);
    int kc = 0;
    for (; kc + 128 <= K; kc += 64) {
        __builtin_prefetch(arow + kc + 128, 0, 1);
        __builtin_prefetch(brow + kc + 128, 0, 1);
        loadAB(a1, b1, kc + 32);
        compute(a0, b0);
        loadAB(a0, b0, kc + 64);
        compute(a1, b1);
    }
    loadAB(a1, b1, kc + 32);
    compute(a0, b0);
    compute(a1, b1);

#pragma unroll
    for (int f = 0; f < 4; ++f) {
        const int   ncol = nBase + f * 16 + l15;
        const float bv   = bias[ncol];
#pragma unroll
        for (int mf = 0; mf < 2; ++mf)
#pragma unroll
            for (int r = 0; r < 8; ++r) {
                const int m = mBase + mf * 16 + r + hhalf * 8;
                out[(size_t)m * EMBED + ncol] = acc[mf][f][r] + bv;
            }
    }
}

// ---------------------------------------------------------------------------
// Host-side launcher
// ---------------------------------------------------------------------------
extern "C" void kernel_launch(void* const* d_in, const int* in_sizes, int n_in,
                              void* d_out, int out_size, void* d_ws, size_t ws_size,
                              hipStream_t stream) {
    (void)in_sizes; (void)n_in; (void)out_size; (void)ws_size;

    const float* x     = (const float*)d_in[0];
    const float* w_qkv = (const float*)d_in[1];
    const float* b_qkv = (const float*)d_in[2];
    const float* w_out = (const float*)d_in[3];
    const float* b_out = (const float*)d_in[4];
    float*       out   = (float*)d_out;

    char*  ws  = (char*)d_ws;
    size_t off = 0;
    auto alloc = [&](size_t bytes) -> void* {
        void* p = ws + off;
        off += (bytes + 255) & ~(size_t)255;
        return p;
    };

    unsigned short* xbf    = (unsigned short*)alloc((size_t)SEQ * EMBED * 2);
    unsigned short* wqkvbf = (unsigned short*)alloc((size_t)3 * EMBED * EMBED * 2);
    unsigned short* woutbf = (unsigned short*)alloc((size_t)EMBED * EMBED * 2);
    unsigned short* qbf    = (unsigned short*)alloc((size_t)HEADS * SEQ * HD * 2);
    unsigned short* kbf    = (unsigned short*)alloc((size_t)HEADS * SEQ * HD * 2);
    unsigned short* vtbf   = (unsigned short*)alloc((size_t)HEADS * SEQ * HD * 2);
    unsigned short* attnbf = (unsigned short*)alloc((size_t)SEQ * EMBED * 2);

    {
        int n = SEQ * EMBED;
        cvt_f32_bf16<<<(n + 255) / 256, 256, 0, stream>>>(x, xbf, n);
        n = 3 * EMBED * EMBED;
        cvt_f32_bf16<<<(n + 255) / 256, 256, 0, stream>>>(w_qkv, wqkvbf, n);
        n = EMBED * EMBED;
        cvt_f32_bf16<<<(n + 255) / 256, 256, 0, stream>>>(w_out, woutbf, n);
    }

    qkv_gemm_wmma<<<dim3(3 * EMBED / 64, SEQ / 256), 256, 0, stream>>>(
        xbf, wqkvbf, b_qkv, qbf, kbf, vtbf);

    attn_wmma<<<dim3(SEQ / 64, HEADS), 128, 0, stream>>>(qbf, kbf, vtbf, attnbf);

    out_gemm_wmma<<<dim3(EMBED / 64, SEQ / 256), 256, 0, stream>>>(
        attnbf, woutbf, b_out, out);
}
